// NnBoard768_29566554865844
// MI455X (gfx1250) — compile-verified, matrix operand order checked by reading, not run on hardware
//
#include <hip/hip_runtime.h>
#include <hip/hip_bf16.h>
#include <cstdint>

typedef _Float16 v16h __attribute__((ext_vector_type(16)));
typedef _Float16 v8h  __attribute__((ext_vector_type(8)));
typedef float    v8f  __attribute__((ext_vector_type(8)));

#define FEAT      768
#define FT_OUT    1024
#define NNZ_PER   32
#define KT        (FEAT / 32)      // 24 K-tiles per row
#define NT        (FT_OUT / 16)    // 64 N-tiles
#define ROWSTRIDE 776              // 768 + 8 halves pad: 388 dwords % 64 banks = 4 -> conflict-free b128

union HF16 { v16h v; v8h h[2]; };

// ---------------------------------------------------------------------------
// Prep: f32 ft_w[1024][768] -> f16, packed per WMMA B-fragment:
// packed[nt][kt][c][lane][e], frag = 512 halves (1 KB) so each wave B-load is
// two coalesced 512B global_load_b128 bursts.
// ---------------------------------------------------------------------------
__global__ void pack_ftw_f16_kernel(const float* __restrict__ w,
                                    _Float16* __restrict__ out, int n) {
  int i = blockIdx.x * blockDim.x + threadIdx.x;
  if (i >= n) return;
  int nt   = i / (KT * 512);
  int r    = i % (KT * 512);
  int kt   = r / 512;
  int r2   = r % 512;
  int c    = r2 >> 8;          // chunk 0/1
  int lane = (r2 >> 3) & 31;
  int e    = r2 & 7;
  int col  = lane & 15;
  int hs   = lane >> 4;
  int o    = nt * 16 + col;                    // output neuron (B-matrix N)
  int k    = kt * 32 + c * 16 + hs * 8 + e;    // feature (B-matrix K)
  out[i] = (_Float16)w[o * FEAT + k];
}

// ---------------------------------------------------------------------------
// Fused NNUE forward: scatter -> WMMA feature transform -> clamp -> bucketed
// output layer -> sigmoid. One workgroup (8 wave32) per 16 boards.
// ---------------------------------------------------------------------------
__global__ __launch_bounds__(256)
void nnue_fused_kernel(const int*   __restrict__ stmIdx,   // [nnz][2]
                       const int*   __restrict__ nstmIdx,  // [nnz][2]
                       const float* __restrict__ values,   // [nnz]
                       const int*   __restrict__ buckets,  // [B]
                       const float* __restrict__ ft_b,     // [1024]
                       const float* __restrict__ out_w,    // [8][2048]
                       const float* __restrict__ out_b,    // [8]
                       const _Float16* __restrict__ ftwp,  // packed f16 weights
                       float* __restrict__ out,            // [B]
                       int nBoards)
{
  __shared__ _Float16 Atile[2][16][ROWSTRIDE];   // stm/nstm board tiles, f16
  __shared__ float    accW[8][16];               // per-wave layer-2 partials
  __shared__ int      bktS[16];

  const int tid      = threadIdx.x;
  const int lane     = tid & 31;
  const int wave     = tid >> 5;
  const int tileBase = blockIdx.x * 16;

  // ---- phase 1: zero A tiles, load buckets -------------------------------
  {
    uint32_t* z = (uint32_t*)&Atile[0][0][0];
    const int nz = 2 * 16 * ROWSTRIDE / 2;       // halves -> dwords
    for (int i = tid; i < nz; i += 256) z[i] = 0u;
    if (tid < 16) {
      int b = tileBase + tid;
      bktS[tid] = (b < nBoards) ? buckets[b] : 0;
    }
  }
  __syncthreads();

  // ---- phase 2: race-free sparse scatter (1 thread per board-row) --------
  if (tid < 32) {
    int persp = tid >> 4;
    int bl    = tid & 15;
    int b     = tileBase + bl;
    if (b < nBoards) {
      const int* idx = persp ? nstmIdx : stmIdx;
      int base = b * NNZ_PER;
      _Float16* row = &Atile[persp][bl][0];
      for (int j = 0; j < NNZ_PER; ++j) {
        int i = base + j;
        int f = idx[2 * i + 1];
        row[f] += (_Float16)values[i];           // duplicates sum serially
      }
    }
  }
  __syncthreads();

  // ---- phase 3: WMMA feature transform + fused output layer --------------
  const int col     = lane & 15;
  const int halfSel = lane >> 4;
  const int k1      = halfSel * 8;               // A chunk offsets (halves)
  const int k2      = 16 + halfSel * 8;
  const _Float16* Arow0 = &Atile[0][col][0];
  const _Float16* Arow1 = &Atile[1][col][0];

  float part[8];
  #pragma unroll
  for (int j = 0; j < 8; ++j) part[j] = 0.0f;

  for (int n = wave; n < NT; n += 8) {
    const int o = n * 16 + col;
    const _Float16* Bn = ftwp + (size_t)n * (KT * 512);
    v8f c0 = {};                                  // stm accumulator
    v8f c1 = {};                                  // nstm accumulator
    #pragma unroll 4
    for (int k = 0; k < KT; ++k) {
      const int kb = k * 32;
      const _Float16* Bk = Bn + k * 512;
      HF16 bf, a0, a1;
      bf.h[0] = *(const v8h*)(Bk + lane * 8);          // coalesced 512B burst
      bf.h[1] = *(const v8h*)(Bk + 256 + lane * 8);
      a0.h[0] = *(const v8h*)(Arow0 + kb + k1);        // ds_load_b128 x2
      a0.h[1] = *(const v8h*)(Arow0 + kb + k2);
      a1.h[0] = *(const v8h*)(Arow1 + kb + k1);
      a1.h[1] = *(const v8h*)(Arow1 + kb + k2);
      c0 = __builtin_amdgcn_wmma_f32_16x16x32_f16(false, a0.v, false, bf.v,
                                                  (short)0, c0, false, false);
      c1 = __builtin_amdgcn_wmma_f32_16x16x32_f16(false, a1.v, false, bf.v,
                                                  (short)0, c1, false, false);
    }
    // epilogue: bias + clamp + bucketed out_w dot, reduced across 16 lanes
    const float bias = ft_b[o];
    #pragma unroll
    for (int j = 0; j < 8; ++j) {
      const int m  = j + (halfSel << 3);          // board within tile
      const int bk = bktS[m];
      float h0 = fminf(fmaxf(c0[j] + bias, 0.0f), 1.0f);
      float h1 = fminf(fmaxf(c1[j] + bias, 0.0f), 1.0f);
      const float* wrow = out_w + bk * (2 * FT_OUT);
      float v = h0 * wrow[o] + h1 * wrow[FT_OUT + o];
      v += __shfl_xor(v, 8, 16);
      v += __shfl_xor(v, 4, 16);
      v += __shfl_xor(v, 2, 16);
      v += __shfl_xor(v, 1, 16);
      if (col == 0) part[j] += v;                 // lanes 0 & 16 hold sums
    }
  }
  // deterministic per-wave partial dump (lanes 0 and 16 cover all 16 boards)
  if (col == 0) {
    #pragma unroll
    for (int j = 0; j < 8; ++j) accW[wave][j + (halfSel << 3)] = part[j];
  }
  __syncthreads();

  // ---- phase 4: fixed-order reduction + sigmoid --------------------------
  if (tid < 16) {
    int b = tileBase + tid;
    if (b < nBoards) {
      float s = 0.0f;
      #pragma unroll
      for (int w = 0; w < 8; ++w) s += accW[w][tid];
      float l1 = s + out_b[bktS[tid]];
      out[b] = 1.0f / (1.0f + __expf(-l1));
    }
  }
}

// ---------------------------------------------------------------------------
extern "C" void kernel_launch(void* const* d_in, const int* in_sizes, int n_in,
                              void* d_out, int out_size, void* d_ws, size_t ws_size,
                              hipStream_t stream) {
  const int*   stmIdx  = (const int*)d_in[0];
  const int*   nstmIdx = (const int*)d_in[1];
  const float* values  = (const float*)d_in[2];
  const int*   buckets = (const int*)d_in[3];
  // d_in[4] = size scalar (unused; sizes derived from in_sizes)
  const float* ft_w    = (const float*)d_in[5];
  const float* ft_b    = (const float*)d_in[6];
  const float* out_w   = (const float*)d_in[7];
  const float* out_b   = (const float*)d_in[8];

  const int nBoards = in_sizes[3];          // 16384
  const int nW      = in_sizes[5];          // 1024*768
  _Float16* ftwp = (_Float16*)d_ws;         // 1.5 MB packed f16 weights

  pack_ftw_f16_kernel<<<(nW + 255) / 256, 256, 0, stream>>>(ft_w, ftwp, nW);

  const int nTiles = (nBoards + 15) / 16;
  nnue_fused_kernel<<<nTiles, 256, 0, stream>>>(
      stmIdx, nstmIdx, values, buckets, ft_b, out_w, out_b, ftwp,
      (float*)d_out, nBoards);
}
